// RallyNet_71425306132714
// MI455X (gfx1250) — matrix operand order repaired and checked by reading, not compiled, not run on hardware
//
#include <hip/hip_runtime.h>

typedef __attribute__((ext_vector_type(16))) _Float16 v16h;
typedef __attribute__((ext_vector_type(8)))  float    v8f;

#define TSTEPS 100
#define NB     4096
#define DIM    32
#define LAT    32
#define CTXD   64
#define HID    128
#define G3     384
#define NTILES 256   // NB / 16

// ---------------------------------------------------------------------------
// math helpers
// ---------------------------------------------------------------------------
__device__ __forceinline__ float sigmf(float x) { return 1.0f / (1.0f + __expf(-x)); }
__device__ __forceinline__ float splusf(float x) {
  return fmaxf(x, 0.0f) + __logf(1.0f + __expf(-fabsf(x)));
}

// ---------------------------------------------------------------------------
// WMMA fragment helpers (wave32, 16x16x32 f16 -> f32)
//
// A 16x32 f16 (M x K), stored row-major [16][ldk] in LDS:
//   lanes 0-15: M=lane, K 0-7 in elems 0-7, K 16-23 in elems 8-15
//   lanes 16-31: M=lane-16, K 8-15 in elems 0-7, K 24-31 in elems 8-15
// (two contiguous 8-half runs per lane -> 2x ds_load_b128)
//
// B operands are weights, staged ONCE into fragment order: per (kslab,ntile)
// tile a contiguous chunk of 32 lanes x 16 halves, so the per-step load is a
// single contiguous 32-byte read per lane (2x ds_load_b128).
// ---------------------------------------------------------------------------
__device__ __forceinline__ v16h loadA16(const _Float16* A, int ldk, int lane, int kOff) {
  const int m  = lane & 15;
  const int kb = (lane >> 4) << 3;       // 0 or 8
  const _Float16* p = A + m * ldk + kOff;
  v16h a;
#pragma unroll
  for (int e = 0; e < 8; ++e) a[e] = p[kb + e];
#pragma unroll
  for (int e = 0; e < 8; ++e) a[8 + e] = p[16 + kb + e];
  return a;
}

__device__ __forceinline__ v16h loadBfrag(const _Float16* frag, int Nt, int lane, int ks, int nt) {
  return *(const v16h*)(frag + (size_t)((ks * Nt + nt) * 32 + lane) * 16);
}

// Stage global weight W [N][Kw] (row-major) into fragment-ordered LDS chunks.
// Fragment element (chunk, lane, e): k = ks*32 + ((e>>3)<<4) + ((lane>>4)<<3) + (e&7),
//                                    n = nt*16 + (lane&15).
__device__ __forceinline__ void stageB(_Float16* dst, const float* __restrict__ W,
                                       int Kw, int N, int tid) {
  const int Nt = N >> 4;
  const int total = Kw * N;
  for (int i = tid; i < total; i += 256) {
    const int e     = i & 15;
    const int l     = (i >> 4) & 31;
    const int chunk = i >> 9;
    const int ks    = chunk / Nt;
    const int nt    = chunk - ks * Nt;
    const int k     = ks * 32 + ((e >> 3) << 4) + ((l >> 4) << 3) + (e & 7);
    const int n     = nt * 16 + (l & 15);
    dst[i] = (_Float16)W[n * Kw + k];
  }
}

__device__ __forceinline__ void storeC32(float* Cm, int ldn, int lane, int nOff, v8f c) {
  const int n  = (lane & 15) + nOff;
  const int mb = (lane >> 4) << 3;       // 0 or 8
#pragma unroll
  for (int v = 0; v < 8; ++v) Cm[(mb + v) * ldn + n] = c[v];
}

#define WMMA_F16(a, b, c) \
  __builtin_amdgcn_wmma_f32_16x16x32_f16(false, (a), false, (b), (short)0, (c), false, false)

// ---------------------------------------------------------------------------
// Kernel 1: GRU encoder scan + context projection.
// Each block owns 16 batch rows for all T steps; h stays in LDS.
// ctx written to workspace as [t][tile][16][64] (coalesced for kernel 2).
// ---------------------------------------------------------------------------
constexpr int K1_F32  = 16 * G3 * 2 + 16 * HID + 2 * G3 + CTXD + 16 * CTXD;          // 16192
constexpr int K1_F16  = 32 * G3 + HID * G3 + HID * CTXD + 16 * HID + 16 * DIM;       // 72192
constexpr int K1_SMEM = K1_F32 * 4 + K1_F16 * 2;                                     // 209152 B

__global__ __launch_bounds__(256)
void rally_gru_enc(const float* __restrict__ xs,
                   const float* __restrict__ Wih, const float* __restrict__ Whh,
                   const float* __restrict__ bih, const float* __restrict__ bhh,
                   const float* __restrict__ encW, const float* __restrict__ encB,
                   float* __restrict__ ctx) {
  extern __shared__ __align__(32) char smem[];
  float* sGI  = (float*)smem;            // [16][384] x@Wih^T
  float* sGH  = sGI + 16 * G3;           // [16][384] h@Whh^T
  float* sH   = sGH + 16 * G3;           // [16][128] fp32 hidden state
  float* sBih = sH + 16 * HID;           // [384]
  float* sBhh = sBih + G3;               // [384]
  float* sEB  = sBhh + G3;               // [64]
  float* sCtx = sEB + CTXD;              // [16][64]
  _Float16* sWih = (_Float16*)(sCtx + 16 * CTXD);  // 24 frag chunks (K=32,N=384)
  _Float16* sWhh = sWih + 32 * G3;                 // 96 frag chunks (K=128,N=384)
  _Float16* sEnc = sWhh + HID * G3;                // 16 frag chunks (K=128,N=64)
  _Float16* sHh  = sEnc + HID * CTXD;              // [16][128]  f16 hidden
  _Float16* sXh  = sHh + 16 * HID;                 // [16][32]   f16 input

  const int tid  = threadIdx.x;
  const int lane = tid & 31;
  const int wid  = tid >> 5;
  const int tile = blockIdx.x;
  const int row0 = tile * 16;

  // Stage weights once into fragment-ordered f16 chunks.
  stageB(sWih, Wih, DIM, G3, tid);
  stageB(sWhh, Whh, HID, G3, tid);
  stageB(sEnc, encW, HID, CTXD, tid);
  for (int i = tid; i < G3; i += 256) { sBih[i] = bih[i]; sBhh[i] = bhh[i]; }
  if (tid < CTXD) sEB[tid] = encB[tid];
  for (int i = tid; i < 16 * HID; i += 256) { sH[i] = 0.0f; sHh[i] = (_Float16)0.0f; }
  __syncthreads();

  for (int t = 0; t < TSTEPS; ++t) {
    // Load x_t tile (coalesced), convert to f16.
    for (int i = tid; i < 16 * DIM; i += 256) {
      int b = i >> 5, d = i & 31;
      sXh[i] = (_Float16)xs[((size_t)t * NB + row0 + b) * DIM + d];
    }
    if (t + 1 < TSTEPS)
      __builtin_prefetch(&xs[((size_t)(t + 1) * NB + row0) * DIM], 0, 1);  // global_prefetch_b8
    __syncthreads();

    // gi = x @ Wih^T   (K=32, one slab; each wave owns 3 N-tiles of 384)
    {
      v16h ax = loadA16(sXh, DIM, lane, 0);
#pragma unroll
      for (int j = 0; j < 3; ++j) {
        const int nt = wid * 3 + j;
        v8f acc = {};
        v16h bw = loadBfrag(sWih, 24, lane, 0, nt);
        acc = WMMA_F16(ax, bw, acc);
        storeC32(sGI, G3, lane, nt * 16, acc);
      }
    }
    // gh = h @ Whh^T   (K=128, 4 slabs)
#pragma unroll
    for (int j = 0; j < 3; ++j) {
      const int nt = wid * 3 + j;
      v8f acc = {};
#pragma unroll
      for (int ks = 0; ks < 4; ++ks) {
        v16h ah = loadA16(sHh, HID, lane, ks * 32);
        v16h bw = loadBfrag(sWhh, 24, lane, ks, nt);
        acc = WMMA_F16(ah, bw, acc);
      }
      storeC32(sGH, G3, lane, nt * 16, acc);
    }
    __syncthreads();

    // GRU gate update (r,z,n order; n uses gh third separately for r*hn).
    for (int i = tid; i < 16 * HID; i += 256) {
      const int b = i >> 7, j = i & 127;
      const float* gi = sGI + b * G3;
      const float* gh = sGH + b * G3;
      float r  = sigmf(gi[j] + sBih[j] + gh[j] + sBhh[j]);
      float z  = sigmf(gi[HID + j] + sBih[HID + j] + gh[HID + j] + sBhh[HID + j]);
      float nn = tanhf(gi[2 * HID + j] + sBih[2 * HID + j] + r * (gh[2 * HID + j] + sBhh[2 * HID + j]));
      float hh = (1.0f - z) * nn + z * sH[i];
      sH[i]  = hh;
      sHh[i] = (_Float16)hh;
    }
    __syncthreads();

    // ctx = h @ encW^T + encB  (waves 0..3 own one 16-col tile each)
    if (wid < 4) {
      v8f acc = {};
#pragma unroll
      for (int ks = 0; ks < 4; ++ks) {
        v16h ah = loadA16(sHh, HID, lane, ks * 32);
        v16h bw = loadBfrag(sEnc, 4, lane, ks, wid);
        acc = WMMA_F16(ah, bw, acc);
      }
      storeC32(sCtx, CTXD, lane, wid * 16, acc);
    }
    __syncthreads();
    for (int i = tid; i < 16 * CTXD; i += 256) {
      int c = i & 63;
      ctx[((size_t)t * NTILES + tile) * (16 * CTXD) + i] = sCtx[i] + sEB[c];
    }
    __syncthreads();
  }
}

// ---------------------------------------------------------------------------
// Kernel 2: posterior z0, SDE Euler-Maruyama scan, output projection, logqp.
// ---------------------------------------------------------------------------
constexpr int K2_F32  = 23248;
constexpr int K2_F16  = 62464;
constexpr int K2_SMEM = K2_F32 * 4 + K2_F16 * 2;   // 217920 B

__global__ __launch_bounds__(256)
void rally_sde(const float* __restrict__ ts, const float* __restrict__ eps0,
               const float* __restrict__ noise,
               const float* __restrict__ qzW, const float* __restrict__ qzB,
               const float* __restrict__ fW1, const float* __restrict__ fb1,
               const float* __restrict__ fW2, const float* __restrict__ fb2,
               const float* __restrict__ fW3, const float* __restrict__ fb3,
               const float* __restrict__ hW1, const float* __restrict__ hb1,
               const float* __restrict__ hW2, const float* __restrict__ hb2,
               const float* __restrict__ hW3, const float* __restrict__ hb3,
               const float* __restrict__ gW1, const float* __restrict__ gb1,
               const float* __restrict__ gW2, const float* __restrict__ gb2,
               const float* __restrict__ prW, const float* __restrict__ prB,
               const float* __restrict__ ctx, float* __restrict__ out) {
  extern __shared__ __align__(32) char smem[];
  float* sGW1 = (float*)smem;            // [32][128]
  float* sGB1 = sGW1 + LAT * HID;        // [32][128]
  float* sGW2 = sGB1 + LAT * HID;        // [32][128]
  float* sQW  = sGW2 + LAT * HID;        // [64][64]
  float* sFb1 = sQW + CTXD * CTXD;       // [128]
  float* sFb2 = sFb1 + HID;              // [128]
  float* sHb1 = sFb2 + HID;              // [128]
  float* sHb2 = sHb1 + HID;              // [128]
  float* sFb3 = sHb2 + HID;              // [32]
  float* sHb3 = sFb3 + LAT;              // [32]
  float* sGb2 = sHb3 + LAT;              // [32]
  float* sQb  = sGb2 + LAT;              // [64]
  float* sPb  = sQb + CTXD;              // [32]
  float* sC   = sPb + LAT;               // [16][64]  ctx tile
  float* sA   = sC + 16 * CTXD;          // [16][128] MLP scratch
  float* sF   = sA + 16 * HID;           // [16][32]  f drift
  float* sHdr = sF + 16 * LAT;           // [16][32]  h drift
  float* sG   = sHdr + 16 * LAT;         // [16][32]  diffusion
  float* sY   = sG + 16 * LAT;           // [16][32]  state y
  float* sUU  = sY + 16 * LAT;           // [16][32]  u*u
  float* sP   = sUU + 16 * LAT;          // [16][32]  projection
  float* sLq  = sP + 16 * LAT;           // [16]
  _Float16* sF1  = (_Float16*)(sLq + 16);   // frag chunks (K=96, N=128)
  _Float16* sF2  = sF1 + 96 * HID;          // frag chunks (K=128,N=128)
  _Float16* sF3  = sF2 + HID * HID;         // frag chunks (K=128,N=32)
  _Float16* sHl1 = sF3 + HID * LAT;         // frag chunks (K=32, N=128)
  _Float16* sHl2 = sHl1 + LAT * HID;        // frag chunks (K=128,N=128)
  _Float16* sHl3 = sHl2 + HID * HID;        // frag chunks (K=128,N=32)
  _Float16* sPrj = sHl3 + HID * LAT;        // frag chunks (K=32, N=32)
  _Float16* sYC  = sPrj + LAT * LAT;        // [16][96]  concat(y, c) f16
  _Float16* sYh  = sYC + 16 * 96;           // [16][32]  y f16
  _Float16* sAh  = sYh + 16 * LAT;          // [16][128] f16 activations

  const int tid  = threadIdx.x;
  const int lane = tid & 31;
  const int wid  = tid >> 5;
  const int tile = blockIdx.x;
  const int row0 = tile * 16;

  // ---- stage weights (f32 -> f16, fragment-ordered) ----
  stageB(sF1,  fW1, 96,  HID, tid);
  stageB(sF2,  fW2, HID, HID, tid);
  stageB(sF3,  fW3, HID, LAT, tid);
  stageB(sHl1, hW1, LAT, HID, tid);
  stageB(sHl2, hW2, HID, HID, tid);
  stageB(sHl3, hW3, HID, LAT, tid);
  stageB(sPrj, prW, LAT, LAT, tid);
  for (int i = tid; i < CTXD * CTXD; i += 256) sQW[i] = qzW[i];
  for (int i = tid; i < LAT * HID; i += 256) { sGW1[i] = gW1[i]; sGB1[i] = gb1[i]; sGW2[i] = gW2[i]; }
  if (tid < HID) { sFb1[tid] = fb1[tid]; sFb2[tid] = fb2[tid]; sHb1[tid] = hb1[tid]; sHb2[tid] = hb2[tid]; }
  if (tid < LAT) { sFb3[tid] = fb3[tid]; sHb3[tid] = hb3[tid]; sGb2[tid] = gb2[tid]; sPb[tid] = prB[tid]; }
  if (tid < CTXD) sQb[tid] = qzB[tid];
  if (tid < 16) sLq[tid] = 0.0f;
  // ctx[0] for posterior
  for (int i = tid; i < 16 * CTXD; i += 256)
    sC[i] = ctx[(size_t)tile * (16 * CTXD) + i];
  __syncthreads();

  // ---- posterior z0 = mean + exp(logstd) * eps0 ----
  for (int o = tid; o < 16 * 2 * LAT; o += 256) {
    int b = o >> 6, q = o & 63;
    float acc = sQb[q];
    for (int c = 0; c < CTXD; ++c) acc = fmaf(sC[b * CTXD + c], sQW[q * CTXD + c], acc);
    sA[o] = acc;
  }
  __syncthreads();
  for (int i = tid; i < 16 * LAT; i += 256) {
    int b = i >> 5, l = i & 31;
    float z = sA[b * 64 + l] + __expf(sA[b * 64 + 32 + l]) * eps0[((size_t)(row0 + b)) * LAT + l];
    sY[i] = z; sYh[i] = (_Float16)z; sYC[b * 96 + l] = (_Float16)z;
  }
  __syncthreads();

  // ---- time loop ----
  for (int t = 0; t < TSTEPS; ++t) {
    // project current y -> xs_hat[t]  (waves 0,1 own the two 16-col tiles)
    if (wid < 2) {
      v8f acc = {};
      v16h ay = loadA16(sYh, LAT, lane, 0);
      v16h bw = loadBfrag(sPrj, 2, lane, 0, wid);
      acc = WMMA_F16(ay, bw, acc);
      storeC32(sP, DIM, lane, wid * 16, acc);
    }
    __syncthreads();
    for (int i = tid; i < 16 * DIM; i += 256) {
      int b = i >> 5, d = i & 31;
      out[((size_t)t * NB + row0 + b) * DIM + d] = sP[i] + sPb[d];
    }
    if (t == TSTEPS - 1) break;

    const float dt = ts[t + 1] - ts[t];
    const float sq = sqrtf(dt);

    // ctx[t] -> sC and the context half of the f-net input
    for (int i = tid; i < 16 * CTXD; i += 256) {
      float c = ctx[((size_t)t * NTILES + tile) * (16 * CTXD) + i];
      sC[i] = c;
      int b = i >> 6, cc = i & 63;
      sYC[b * 96 + 32 + cc] = (_Float16)c;
    }
    __syncthreads();

    // ---- f_mlp: softplus(softplus([y,c]W1+b1)W2+b2)W3+b3 ----
    {
      v8f acc = {};
#pragma unroll
      for (int ks = 0; ks < 3; ++ks) {
        v16h a = loadA16(sYC, 96, lane, ks * 32);
        v16h b = loadBfrag(sF1, 8, lane, ks, wid);
        acc = WMMA_F16(a, b, acc);
      }
      storeC32(sA, HID, lane, wid * 16, acc);
    }
    __syncthreads();
    for (int i = tid; i < 16 * HID; i += 256) sAh[i] = (_Float16)splusf(sA[i] + sFb1[i & 127]);
    __syncthreads();
    {
      v8f acc = {};
#pragma unroll
      for (int ks = 0; ks < 4; ++ks) {
        v16h a = loadA16(sAh, HID, lane, ks * 32);
        v16h b = loadBfrag(sF2, 8, lane, ks, wid);
        acc = WMMA_F16(a, b, acc);
      }
      storeC32(sA, HID, lane, wid * 16, acc);
    }
    __syncthreads();
    for (int i = tid; i < 16 * HID; i += 256) sAh[i] = (_Float16)splusf(sA[i] + sFb2[i & 127]);
    __syncthreads();
    if (wid < 2) {
      v8f acc = {};
#pragma unroll
      for (int ks = 0; ks < 4; ++ks) {
        v16h a = loadA16(sAh, HID, lane, ks * 32);
        v16h b = loadBfrag(sF3, 2, lane, ks, wid);
        acc = WMMA_F16(a, b, acc);
      }
      storeC32(sF, LAT, lane, wid * 16, acc);
    }
    __syncthreads();

    // ---- h_mlp (prior drift) ----
    {
      v8f acc = {};
      v16h a = loadA16(sYh, LAT, lane, 0);
      v16h b = loadBfrag(sHl1, 8, lane, 0, wid);
      acc = WMMA_F16(a, b, acc);
      storeC32(sA, HID, lane, wid * 16, acc);
    }
    __syncthreads();
    for (int i = tid; i < 16 * HID; i += 256) sAh[i] = (_Float16)splusf(sA[i] + sHb1[i & 127]);
    __syncthreads();
    {
      v8f acc = {};
#pragma unroll
      for (int ks = 0; ks < 4; ++ks) {
        v16h a = loadA16(sAh, HID, lane, ks * 32);
        v16h b = loadBfrag(sHl2, 8, lane, ks, wid);
        acc = WMMA_F16(a, b, acc);
      }
      storeC32(sA, HID, lane, wid * 16, acc);
    }
    __syncthreads();
    for (int i = tid; i < 16 * HID; i += 256) sAh[i] = (_Float16)splusf(sA[i] + sHb2[i & 127]);
    __syncthreads();
    if (wid < 2) {
      v8f acc = {};
#pragma unroll
      for (int ks = 0; ks < 4; ++ks) {
        v16h a = loadA16(sAh, HID, lane, ks * 32);
        v16h b = loadBfrag(sHl3, 2, lane, ks, wid);
        acc = WMMA_F16(a, b, acc);
      }
      storeC32(sHdr, LAT, lane, wid * 16, acc);
    }
    __syncthreads();

    // ---- diag g nets + KL term + Euler-Maruyama update ----
    for (int i = tid; i < 16 * LAT; i += 256) {
      const int b = i >> 5, l = i & 31;
      const float y = sY[i];
      const float* w1 = sGW1 + l * HID;
      const float* b1 = sGB1 + l * HID;
      const float* w2 = sGW2 + l * HID;
      float acc = 0.0f;
      for (int hh = 0; hh < HID; ++hh)
        acc += splusf(fmaf(y, w1[hh], b1[hh])) * w2[hh];
      const float g = sigmf(acc + sGb2[l]);
      sG[i] = g;
      const float f  = sF[i] + sFb3[l];
      const float hd = sHdr[i] + sHb3[l];
      const float u  = (f - hd) / g;
      sUU[i] = u * u;
      const float dW = noise[((size_t)t * NB + row0 + b) * LAT + l];
      const float yn = y + f * dt + g * sq * dW;
      sY[i] = yn; sYh[i] = (_Float16)yn; sYC[b * 96 + l] = (_Float16)yn;
    }
    __syncthreads();
    if (tid < 16) {
      float s = 0.0f;
      for (int l = 0; l < LAT; ++l) s += sUU[tid * LAT + l];
      sLq[tid] += 0.5f * s * dt;
    }
    __syncthreads();
  }

  if (tid < 16) out[(size_t)TSTEPS * NB * DIM + row0 + tid] = sLq[tid];
}

// ---------------------------------------------------------------------------
extern "C" void kernel_launch(void* const* d_in, const int* in_sizes, int n_in,
                              void* d_out, int out_size, void* d_ws, size_t ws_size,
                              hipStream_t stream) {
  (void)in_sizes; (void)n_in; (void)out_size; (void)ws_size;
  const float* xs   = (const float*)d_in[0];
  const float* ts   = (const float*)d_in[1];
  const float* eps0 = (const float*)d_in[2];
  const float* noise= (const float*)d_in[3];
  const float* Wih  = (const float*)d_in[4];
  const float* Whh  = (const float*)d_in[5];
  const float* bih  = (const float*)d_in[6];
  const float* bhh  = (const float*)d_in[7];
  const float* encW = (const float*)d_in[8];
  const float* encB = (const float*)d_in[9];
  const float* qzW  = (const float*)d_in[10];
  const float* qzB  = (const float*)d_in[11];
  const float* fW1  = (const float*)d_in[12];
  const float* fb1  = (const float*)d_in[13];
  const float* fW2  = (const float*)d_in[14];
  const float* fb2  = (const float*)d_in[15];
  const float* fW3  = (const float*)d_in[16];
  const float* fb3  = (const float*)d_in[17];
  const float* hW1  = (const float*)d_in[18];
  const float* hb1  = (const float*)d_in[19];
  const float* hW2  = (const float*)d_in[20];
  const float* hb2  = (const float*)d_in[21];
  const float* hW3  = (const float*)d_in[22];
  const float* hb3  = (const float*)d_in[23];
  const float* gW1  = (const float*)d_in[24];
  const float* gb1  = (const float*)d_in[25];
  const float* gW2  = (const float*)d_in[26];
  const float* gb2  = (const float*)d_in[27];
  const float* prW  = (const float*)d_in[28];
  const float* prB  = (const float*)d_in[29];

  float* ctx = (float*)d_ws;   // [T][NTILES][16][CTXD] fp32 = 100 MB

  rally_gru_enc<<<dim3(NTILES), dim3(256), K1_SMEM, stream>>>(
      xs, Wih, Whh, bih, bhh, encW, encB, ctx);
  rally_sde<<<dim3(NTILES), dim3(256), K2_SMEM, stream>>>(
      ts, eps0, noise, qzW, qzB,
      fW1, fb1, fW2, fb2, fW3, fb3,
      hW1, hb1, hW2, hb2, hW3, hb3,
      gW1, gb1, gW2, gb2, prW, prB,
      ctx, (float*)d_out);
}
